// MambaEncoder_1211180777596
// MI455X (gfx1250) — compile-verified, matrix operand order checked by reading, not run on hardware
//
#include <hip/hip_runtime.h>

// ---------------------------------------------------------------------------
// Model dims (compile-time constants from the reference)
// ---------------------------------------------------------------------------
#define DMODEL 768
#define DINNER 1536
#define DSTATE 16
#define DCONV  4
#define DTRANK 48
#define DTPAD  64          // DT_RANK padded to a multiple of 64 for WMMA K
#define NDEPTH 4
#define FFDIM  3072
#define NBATCH 2
#define SEQL   1024
#define BLTOK  (NBATCH * SEQL)        // 2048 token rows
#define XDBLW  (DTRANK + 2 * DSTATE)  // 80

// GEMM tiling
#define BM 128
#define BN 128
#define BK 64
#define LDP 80   // LDS row pitch in halfs (160B: 16B aligned, 40-bank stride)

// ---------------------------------------------------------------------------
// WMMA types (gfx1250: wave32, 16x16x32 bf16 -> f32)
// ---------------------------------------------------------------------------
typedef __attribute__((ext_vector_type(16))) __bf16 v16bf;
typedef __attribute__((ext_vector_type(8)))  float  v8f;

union FragU { uint4 u[2]; v16bf v; };

__device__ __forceinline__ v16bf frag_load(const unsigned short* p) {
    FragU f;
    f.u[0] = *(const uint4*)(p);
    f.u[1] = *(const uint4*)(p + 8);
    return f.v;
}

__device__ __forceinline__ unsigned short f32_to_bf16(float f) {
    unsigned int u = __float_as_uint(f);
    unsigned int r = (u + 0x7FFFu + ((u >> 16) & 1u)) >> 16;   // RNE
    return (unsigned short)r;
}

__device__ __forceinline__ float gelu_f(float x)     { return 0.5f * x * (1.0f + erff(x * 0.70710678118654752f)); }
__device__ __forceinline__ float softplus_f(float x) { return (x > 20.0f) ? x : log1pf(expf(x)); }
__device__ __forceinline__ float silu_f(float x)     { return x / (1.0f + expf(-x)); }

enum { EPI_NONE = 0, EPI_BIAS = 1, EPI_GELU = 2, EPI_SOFTPLUS = 3 };

// Async global -> LDS copy, 16 bytes per lane (CDNA5, tracked by ASYNCcnt).
// lds_off = LDS byte offset (low 32 bits of a shared-aperture flat address).
__device__ __forceinline__ void async_ld_b128(unsigned lds_off, const unsigned short* g) {
    asm volatile("global_load_async_to_lds_b128 %0, %1, off"
                 :: "v"(lds_off), "v"(g) : "memory");
}
// Async loads complete in order: waiting for <=N leaves the newest N in flight.
__device__ __forceinline__ void wait_asynccnt0() {
    asm volatile("s_wait_asynccnt 0" ::: "memory");
}
__device__ __forceinline__ void wait_asynccnt4() {
    asm volatile("s_wait_asynccnt 4" ::: "memory");
}

// ---------------------------------------------------------------------------
// GEMM: C[M,N] = A[M,K] * W[N,K]^T  (both bf16 row-major with contiguous K)
// Block = 256 threads = 8 wave32s; block tile 128x128; wave tile 64x32.
// K pipelined in 64-wide stages through double-buffered LDS via
// global_load_async_to_lds_b128; stage s+1 copy overlaps stage s WMMAs.
// Per K-step each wave runs 16 v_wmma_f32_16x16x32_bf16.
// Requires: K % 64 == 0, N % 16 == 0 (guaranteed by callers; DTRANK padded).
// ---------------------------------------------------------------------------
__global__ __launch_bounds__(256)
void me_gemm_bf16_nt(const unsigned short* __restrict__ A,
                     const unsigned short* __restrict__ W,
                     float* __restrict__ C,
                     const float* __restrict__ bias,
                     int M, int N, int K,
                     int lda, int ldw, int ldc, int epi)
{
    __shared__ unsigned short ldsA[2][BM * LDP];
    __shared__ unsigned short ldsB[2][BN * LDP];

    const int tid  = threadIdx.x;
    const int lane = tid & 31;
    const int wave = tid >> 5;
    const int wm = wave >> 2;            // 0..1  (M)
    const int wn = wave & 3;             // 0..3  (N)
    const int m0b = blockIdx.y * BM;
    const int n0b = blockIdx.x * BN;
    const int m0 = wm * 64;              // wave tile origin within block tile
    const int n0 = wn * 32;
    const int fr = lane & 15;            // fragment row
    const int kh = (lane >> 4) << 4;     // K half: 0 or 16

    // Loader assignment: each tile = 128 rows x 4 chunks(16B); 512 chunks,
    // 2 per thread per tile -> 4 async ops per thread per stage.
    int lrow[2], lcc[2], garow[2], gbrow[2];
#pragma unroll
    for (int i = 0; i < 2; ++i) {
        const int chunk = tid + 256 * i;     // 0..511
        lrow[i] = chunk >> 2;                // 0..127
        lcc[i]  = (chunk & 3) << 4;          // 0,16,32,48 (halfs)
        int ga = m0b + lrow[i]; garow[i] = (ga > M - 1) ? (M - 1) : ga;  // clamp;
        int gb = n0b + lrow[i]; gbrow[i] = (gb > N - 1) ? (N - 1) : gb;  // stores guarded
    }

    v8f acc[4][2];
#pragma unroll
    for (int i = 0; i < 4; ++i)
#pragma unroll
        for (int j = 0; j < 2; ++j)
            acc[i][j] = (v8f){0.f, 0.f, 0.f, 0.f, 0.f, 0.f, 0.f, 0.f};

    // Issue one stage of async global->LDS copies into buffer `buf`.
    auto stage = [&](int k0, int buf) {
#pragma unroll
        for (int i = 0; i < 2; ++i) {
            const int row = lrow[i], cc = lcc[i];
            async_ld_b128((unsigned)(size_t)&ldsA[buf][row * LDP + cc],
                          A + (size_t)garow[i] * lda + k0 + cc);
            async_ld_b128((unsigned)(size_t)&ldsB[buf][row * LDP + cc],
                          W + (size_t)gbrow[i] * ldw + k0 + cc);
        }
    };

    const int nsteps = K / BK;
    stage(0, 0);

    for (int s = 0; s < nsteps; ++s) {
        const int cur = s & 1;
        const bool more = (s + 1) < nsteps;
        if (more) stage((s + 1) * BK, cur ^ 1);   // prefetch next stage
        if (more) wait_asynccnt4();               // stage s done (in-order), s+1 in flight
        else      wait_asynccnt0();
        __syncthreads();                          // stage-s tiles visible to all waves

        const unsigned short* tA = ldsA[cur];
        const unsigned short* tB = ldsB[cur];
#pragma unroll
        for (int kc = 0; kc < 2; ++kc) {
            const int kk = kc * 32 + kh;
            v16bf af[4], bf[2];
#pragma unroll
            for (int i = 0; i < 4; ++i)
                af[i] = frag_load(&tA[(m0 + 16 * i + fr) * LDP + kk]);
#pragma unroll
            for (int j = 0; j < 2; ++j)
                bf[j] = frag_load(&tB[(n0 + 16 * j + fr) * LDP + kk]);
#pragma unroll
            for (int i = 0; i < 4; ++i)
#pragma unroll
                for (int j = 0; j < 2; ++j)
                    acc[i][j] = __builtin_amdgcn_wmma_f32_16x16x32_bf16(
                        false, af[i], false, bf[j], (short)0, acc[i][j], false, false);
        }
        __syncthreads();   // buf[cur] consumed; safe to overwrite at stage s+2
    }

    // C/D layout: lane 0..15 -> N=lane, M=vgpr; lane 16..31 -> N=lane-16, M=8+vgpr
    const int colb = fr;
    const int rowsel = (lane >> 4) << 3;
#pragma unroll
    for (int j = 0; j < 2; ++j) {
        const int col = n0b + n0 + 16 * j + colb;
        if (col >= N) continue;
        const float bv = bias ? bias[col] : 0.0f;
#pragma unroll
        for (int i = 0; i < 4; ++i) {
#pragma unroll
            for (int v = 0; v < 8; ++v) {
                const int row = m0b + m0 + 16 * i + rowsel + v;
                if (row >= M) continue;
                float x = acc[i][j][v];
                if      (epi == EPI_BIAS)     x += bv;
                else if (epi == EPI_GELU)     x = gelu_f(x + bv);
                else if (epi == EPI_SOFTPLUS) x = softplus_f(x + bv);
                C[(size_t)row * ldc + col] = x;
            }
        }
    }
}

// ---------------------------------------------------------------------------
// f32 -> bf16 pack with zero padding (cols..dld-1) and optional L-flip
// ---------------------------------------------------------------------------
__global__ __launch_bounds__(256)
void me_pack_bf16(const float* __restrict__ src, unsigned short* __restrict__ dst,
                  int rows, int cols, int sld, int dld, int flipL)
{
    int idx = blockIdx.x * 256 + threadIdx.x;
    if (idx >= rows * dld) return;
    int r = idx / dld, c = idx - r * dld;
    int sr = r;
    if (flipL) {
        int b = r / SEQL, l = r - b * SEQL;
        sr = b * SEQL + (SEQL - 1 - l);
    }
    float v = (c < cols) ? src[(size_t)sr * sld + c] : 0.0f;
    dst[idx] = f32_to_bf16(v);
}

// ---------------------------------------------------------------------------
// LayerNorm over last dim (cols), one block per row
// ---------------------------------------------------------------------------
__global__ __launch_bounds__(256)
void me_layernorm(const float* __restrict__ in, float* __restrict__ out,
                  const float* __restrict__ w, const float* __restrict__ b, int cols)
{
    __shared__ float s1[256], s2[256];
    const int row = blockIdx.x, tid = threadIdx.x;
    const float* p = in + (size_t)row * cols;
    float s = 0.f, q = 0.f;
    for (int c = tid; c < cols; c += 256) { float v = p[c]; s += v; q += v * v; }
    s1[tid] = s; s2[tid] = q;
    __syncthreads();
    for (int st = 128; st > 0; st >>= 1) {
        if (tid < st) { s1[tid] += s1[tid + st]; s2[tid] += s2[tid + st]; }
        __syncthreads();
    }
    const float mean = s1[0] / cols;
    const float var  = s2[0] / cols - mean * mean;
    const float inv  = rsqrtf(var + 1e-5f);
    float* o = out + (size_t)row * cols;
    for (int c = tid; c < cols; c += 256) o[c] = (p[c] - mean) * inv * w[c] + b[c];
}

// ---------------------------------------------------------------------------
// Depthwise causal conv1d (K=4) on xc = xz[:, :, :DINNER], + bias, SiLU
// ---------------------------------------------------------------------------
__global__ __launch_bounds__(256)
void me_conv_silu(const float* __restrict__ xz, const float* __restrict__ cw,
                  const float* __restrict__ cb, float* __restrict__ xc)
{
    int idx = blockIdx.x * 256 + threadIdx.x;
    if (idx >= BLTOK * DINNER) return;
    int c = idx % DINNER;
    int t = idx / DINNER;
    int l = t % SEQL, b = t / SEQL;
    float acc = cb[c];
#pragma unroll
    for (int k = 0; k < DCONV; ++k) {
        int ls = l - (DCONV - 1) + k;
        if (ls >= 0)
            acc += cw[c * DCONV + k] * xz[(size_t)(b * SEQL + ls) * (2 * DINNER) + c];
    }
    xc[(size_t)t * DINNER + c] = silu_f(acc);
}

// ---------------------------------------------------------------------------
// Selective scan, fused with +x*D and *silu(z). One thread per (b, e);
// 16-state recurrence kept in registers, sequential over L.
// ---------------------------------------------------------------------------
__global__ __launch_bounds__(256)
void me_scan(const float* __restrict__ xz, const float* __restrict__ xc,
             const float* __restrict__ dt, const float* __restrict__ xdbl,
             const float* __restrict__ A_log, const float* __restrict__ Dp,
             float* __restrict__ y)
{
    int idx = blockIdx.x * 256 + threadIdx.x;
    if (idx >= NBATCH * DINNER) return;
    int e = idx % DINNER, b = idx / DINNER;

    float Arow[DSTATE], h[DSTATE];
#pragma unroll
    for (int s = 0; s < DSTATE; ++s) { Arow[s] = -expf(A_log[e * DSTATE + s]); h[s] = 0.f; }
    const float Dv = Dp[e];

    for (int l = 0; l < SEQL; ++l) {
        const size_t t = (size_t)(b * SEQL + l);
        const float dtv = dt[t * DINNER + e];
        const float xv  = xc[t * DINNER + e];
        const float zv  = xz[t * (2 * DINNER) + DINNER + e];
        const float* bc = xdbl + t * XDBLW;
        const float dx = dtv * xv;
        float accv = 0.f;
#pragma unroll
        for (int s = 0; s < DSTATE; ++s) {
            const float dA = __expf(dtv * Arow[s]);
            h[s] = dA * h[s] + dx * bc[DTRANK + s];
            accv += h[s] * bc[DTRANK + DSTATE + s];
        }
        y[t * DINNER + e] = (accv + xv * Dv) * silu_f(zv);
    }
}

// ---------------------------------------------------------------------------
// Direction combine: flip==0 -> dst = src; flip==1 -> dst = 0.5*(dst + flipL(src))
// ---------------------------------------------------------------------------
__global__ __launch_bounds__(256)
void me_combine(float* __restrict__ dst, const float* __restrict__ src, int flip)
{
    int idx = blockIdx.x * 256 + threadIdx.x;
    if (idx >= BLTOK * DMODEL) return;
    if (!flip) { dst[idx] = src[idx]; return; }
    int r = idx / DMODEL, c = idx - r * DMODEL;
    int b = r / SEQL, l = r - b * SEQL;
    float sv = src[(size_t)(b * SEQL + (SEQL - 1 - l)) * DMODEL + c];
    dst[idx] = 0.5f * (dst[idx] + sv);
}

__global__ __launch_bounds__(256)
void me_add_inplace(float* __restrict__ a, const float* __restrict__ b, int n)
{
    int idx = blockIdx.x * 256 + threadIdx.x;
    if (idx < n) a[idx] += b[idx];
}

__global__ __launch_bounds__(256)
void me_zero(float* __restrict__ a, int n)
{
    int idx = blockIdx.x * 256 + threadIdx.x;
    if (idx < n) a[idx] = 0.0f;
}

// ---------------------------------------------------------------------------
// Host orchestration
// ---------------------------------------------------------------------------
extern "C" void kernel_launch(void* const* d_in, const int* in_sizes, int n_in,
                              void* d_out, int out_size, void* d_ws, size_t ws_size,
                              hipStream_t stream)
{
    (void)in_sizes; (void)n_in; (void)out_size; (void)ws_size;

    // ---- workspace carving (256B aligned) --------------------------------
    size_t off = 0;
    auto carve = [&](size_t bytes) -> void* {
        off = (off + 255) & ~(size_t)255;
        void* p = (char*)d_ws + off;
        off += bytes;
        return p;
    };
    float* resid    = (float*)carve((size_t)BLTOK * DMODEL * 4);
    float* xcur     = (float*)carve((size_t)BLTOK * DMODEL * 4);
    float* hnorm    = (float*)carve((size_t)BLTOK * DMODEL * 4);
    float* tmp768   = (float*)carve((size_t)BLTOK * DMODEL * 4);
    float* mout     = (float*)carve((size_t)BLTOK * DMODEL * 4);
    float* buf3072  = (float*)carve((size_t)BLTOK * (2 * DINNER) * 4); // xz / MLP hidden
    float* xcbuf    = (float*)carve((size_t)BLTOK * DINNER * 4);
    float* dtbuf    = (float*)carve((size_t)BLTOK * DINNER * 4);
    float* ybuf     = (float*)carve((size_t)BLTOK * DINNER * 4);
    float* xdbl     = (float*)carve((size_t)BLTOK * XDBLW * 4);
    unsigned short* actbf = (unsigned short*)carve((size_t)BLTOK * (2 * DINNER) * 2);
    unsigned short* wbf   = (unsigned short*)carve((size_t)(2 * DINNER) * DMODEL * 2);

    auto IN = [&](int i) -> const float* { return (const float*)d_in[i]; };

    auto pack = [&](const float* src, unsigned short* dst, int rows, int cols,
                    int sld, int dld, int flip) {
        int n = rows * dld;
        me_pack_bf16<<<(n + 255) / 256, 256, 0, stream>>>(src, dst, rows, cols, sld, dld, flip);
    };
    auto gemm = [&](const unsigned short* A, const unsigned short* W, float* C,
                    const float* bias, int M, int N, int K, int lda, int ldw, int ldc, int epi) {
        dim3 g((N + BN - 1) / BN, (M + BM - 1) / BM);
        me_gemm_bf16_nt<<<g, 256, 0, stream>>>(A, W, C, bias, M, N, K, lda, ldw, ldc, epi);
    };
    auto add_ip = [&](float* a, const float* b, int n) {
        me_add_inplace<<<(n + 255) / 256, 256, 0, stream>>>(a, b, n);
    };

    // One Mamba direction. mbase = index of in_proj_w leaf for this mamba.
    // flip=1 -> input is L-flipped at pack time, output flipped back at combine.
    auto mamba_dir = [&](int mbase, int flip) {
        const float* in_proj = IN(mbase + 0);   // (2*DINNER, DMODEL)
        const float* conv_w  = IN(mbase + 1);   // (DINNER, 1, DCONV)
        const float* conv_b  = IN(mbase + 2);   // (DINNER)
        const float* x_proj  = IN(mbase + 3);   // (XDBLW, DINNER)
        const float* dt_w    = IN(mbase + 4);   // (DINNER, DTRANK)
        const float* dt_b    = IN(mbase + 5);   // (DINNER)
        const float* A_log   = IN(mbase + 6);   // (DINNER, DSTATE)
        const float* Dp      = IN(mbase + 7);   // (DINNER)
        const float* out_w   = IN(mbase + 8);   // (DMODEL, DINNER)

        // in_proj: xz = h @ in_proj^T      (2048 x 3072)
        pack(hnorm, actbf, BLTOK, DMODEL, DMODEL, DMODEL, flip);
        pack(in_proj, wbf, 2 * DINNER, DMODEL, DMODEL, DMODEL, 0);
        gemm(actbf, wbf, buf3072, nullptr, BLTOK, 2 * DINNER, DMODEL,
             DMODEL, DMODEL, 2 * DINNER, EPI_NONE);

        // depthwise causal conv + bias + silu -> xc
        me_conv_silu<<<(BLTOK * DINNER + 255) / 256, 256, 0, stream>>>(buf3072, conv_w, conv_b, xcbuf);

        // x_proj: x_dbl = xc @ x_proj^T    (2048 x 80)
        pack(xcbuf, actbf, BLTOK, DINNER, DINNER, DINNER, 0);
        pack(x_proj, wbf, XDBLW, DINNER, DINNER, DINNER, 0);
        gemm(actbf, wbf, xdbl, nullptr, BLTOK, XDBLW, DINNER,
             DINNER, DINNER, XDBLW, EPI_NONE);

        // dt_proj (K=48 padded to 64) + bias + softplus -> dt
        pack(xdbl, actbf, BLTOK, DTRANK, XDBLW, DTPAD, 0);
        pack(dt_w, wbf, DINNER, DTRANK, DTRANK, DTPAD, 0);
        gemm(actbf, wbf, dtbuf, dt_b, BLTOK, DINNER, DTPAD,
             DTPAD, DTPAD, DINNER, EPI_SOFTPLUS);

        // selective scan fused with x*D and silu(z)
        me_scan<<<(NBATCH * DINNER + 255) / 256, 256, 0, stream>>>(
            buf3072, xcbuf, dtbuf, xdbl, A_log, Dp, ybuf);

        // out_proj: tmp = y @ out_proj^T   (2048 x 768)
        pack(ybuf, actbf, BLTOK, DINNER, DINNER, DINNER, 0);
        pack(out_w, wbf, DMODEL, DINNER, DINNER, DINNER, 0);
        gemm(actbf, wbf, tmp768, nullptr, BLTOK, DMODEL, DINNER,
             DINNER, DINNER, DMODEL, EPI_NONE);

        me_combine<<<(BLTOK * DMODEL + 255) / 256, 256, 0, stream>>>(mout, tmp768, flip);
    };

    const int NTOKF = BLTOK * DMODEL;

    // x -> xcur, resid = 0
    hipMemcpyAsync(xcur, d_in[0], (size_t)NTOKF * 4, hipMemcpyDeviceToDevice, stream);
    me_zero<<<(NTOKF + 255) / 256, 256, 0, stream>>>(resid, NTOKF);

    // Leaf order (insertion order of setup_inputs dicts):
    //   d_in[0] = x; per block (26 leaves): f_mamba[9], r_mamba[9],
    //   ln1_w, ln1_b, ln2_w, ln2_b, fc1_w, fc1_b, fc2_w, fc2_b
    for (int blk = 0; blk < NDEPTH; ++blk) {
        const int base = 1 + blk * 26;

        // residual = x (+ residual)
        add_ip(resid, xcur, NTOKF);

        // h = LN1(residual); h = bimamba(h)
        me_layernorm<<<BLTOK, 256, 0, stream>>>(resid, hnorm, IN(base + 18), IN(base + 19), DMODEL);
        mamba_dir(base + 0, /*flip=*/0);   // forward
        mamba_dir(base + 9, /*flip=*/1);   // reverse

        // residual = h + residual
        add_ip(resid, mout, NTOKF);

        // h = LN2(residual); x = MLP(h)
        me_layernorm<<<BLTOK, 256, 0, stream>>>(resid, hnorm, IN(base + 20), IN(base + 21), DMODEL);
        pack(hnorm, actbf, BLTOK, DMODEL, DMODEL, DMODEL, 0);
        pack(IN(base + 22), wbf, FFDIM, DMODEL, DMODEL, DMODEL, 0);
        gemm(actbf, wbf, buf3072, IN(base + 23), BLTOK, FFDIM, DMODEL,
             DMODEL, DMODEL, FFDIM, EPI_GELU);
        pack(buf3072, actbf, BLTOK, FFDIM, FFDIM, FFDIM, 0);
        pack(IN(base + 24), wbf, DMODEL, FFDIM, FFDIM, FFDIM, 0);
        gemm(actbf, wbf, xcur, IN(base + 25), BLTOK, DMODEL, FFDIM,
             FFDIM, FFDIM, DMODEL, EPI_BIAS);
    }

    hipMemcpyAsync(d_out, xcur, (size_t)NTOKF * 4, hipMemcpyDeviceToDevice, stream);
}
